// LSTM_62672162783551
// MI455X (gfx1250) — matrix operand order repaired and checked by reading, not compiled, not run on hardware
//
#include <hip/hip_runtime.h>
#include <hip/hip_bf16.h>

// ---------------------------------------------------------------------------
// BiLSTM-CRF for MI455X (gfx1250). bf16 WMMA (16x16x32, f32 accum) for all
// matmuls; f32 scalar math for CRF / gates / MLP. Workspace need ~98MB.
// ---------------------------------------------------------------------------

#define T_LEN 2048
#define EMB_D 50
#define EMB_P 64
#define HDIM 512     // H
#define GDIM 2048    // 4H
#define HID2 1024    // 2H
#define NTAG 16

typedef __attribute__((ext_vector_type(16))) __bf16        v16bf;
typedef __attribute__((ext_vector_type(8)))  float         v8f;
typedef __attribute__((ext_vector_type(4)))  unsigned int  uint4v;

// ---- bf16 helpers (raw u16 storage; round-to-nearest-even encode) ---------
__device__ __forceinline__ unsigned short f2bf(float f) {
    union { float f; unsigned int u; } v; v.f = f;
    unsigned int u = v.u + 0x7FFFu + ((v.u >> 16) & 1u);
    return (unsigned short)(u >> 16);
}
__device__ __forceinline__ float bf2f(unsigned short h) {
    union { unsigned int u; float f; } v; v.u = ((unsigned int)h) << 16;
    return v.f;
}

// Build a 16x__bf16 fragment from two 16B chunks: chunk0 at p, chunk1 at p+off2
// (off2 = 16 elements for both A- and B-layout; see ISA 7.12.2).
__device__ __forceinline__ v16bf load_frag(const unsigned short* p, int off2) {
    union { v16bf v; uint4v q[2]; } f;
    f.q[0] = *(const uint4v*)(p);
    f.q[1] = *(const uint4v*)(p + off2);
    return f.v;
}

__device__ __forceinline__ v8f wmma_bf16(v16bf a, v16bf b, v8f c) {
    return __builtin_amdgcn_wmma_f32_16x16x32_bf16(false, a, false, b,
                                                   (short)0, c, false, false);
}

__device__ __forceinline__ float sigmf(float x) {
    return 1.0f / (1.0f + __expf(-x));
}

// ---------------------------------------------------------------------------
// f32 -> bf16 convert with optional column padding (ksrc -> kdst, zero fill)
// ---------------------------------------------------------------------------
__global__ void cvt_pad_kernel(const float* __restrict__ src,
                               unsigned short* __restrict__ dst,
                               int rows, int ksrc, int kdst) {
    int i = blockIdx.x * blockDim.x + threadIdx.x;
    if (i >= rows * kdst) return;
    int r = i / kdst, c = i % kdst;
    float v = (c < ksrc) ? src[(size_t)r * ksrc + c] : 0.0f;
    dst[i] = f2bf(v);
}

// ---------------------------------------------------------------------------
// Embedding gather: xb[sent][t][64] (bf16, cols 50..63 zero)
// ---------------------------------------------------------------------------
__global__ void embed_kernel(const int* __restrict__ s1, const int* __restrict__ s2,
                             const float* __restrict__ emb,
                             unsigned short* __restrict__ xb) {
    int i = blockIdx.x * blockDim.x + threadIdx.x;
    if (i >= 2 * T_LEN * EMB_P) return;
    int sent = i / (T_LEN * EMB_P);
    int rem  = i % (T_LEN * EMB_P);
    int t = rem / EMB_P, c = rem % EMB_P;
    const int* s = sent ? s2 : s1;
    float v = 0.0f;
    if (c < EMB_D) v = emb[(size_t)s[t] * EMB_D + c];
    xb[i] = f2bf(v);
}

// ---------------------------------------------------------------------------
// Input-projection GEMM: C[t][j] = sum_k A[t][k] * B[j][k]   (M=T, N=2048)
// A bf16 row-major [T][lda] (sent-selected); B bf16 row-major [2048][ldb]
// (dir-selected). Each wave computes 4 adjacent 16x16 N-tiles sharing one A
// fragment per k-step. The k loop is unrolled x2 only, to bound the
// scheduler's load-hoisting window and avoid VGPR spills.
// ---------------------------------------------------------------------------
template <int KSTEPS>
__global__ __launch_bounds__(256)
void gemm_xg_kernel(const unsigned short* __restrict__ A_base, size_t a_sent_stride, int lda,
                    const unsigned short* __restrict__ B_base, size_t b_dir_stride, int ldb,
                    unsigned short* __restrict__ C_base, size_t c_chain_stride) {
    constexpr int NT = 4;
    const int chain = blockIdx.z;
    const int sent = chain >> 1, dir = chain & 1;
    const unsigned short* A = A_base + (size_t)sent * a_sent_stride;
    const unsigned short* B = B_base + (size_t)dir  * b_dir_stride;
    unsigned short*       C = C_base + (size_t)chain * c_chain_stride;

    const int lane = threadIdx.x & 31;
    const int wave = threadIdx.x >> 5;
    const int l15  = lane & 15;
    const int hi   = (lane >> 4) & 1;

    const int mt     = blockIdx.x;
    const int ntile0 = (blockIdx.y * 8 + wave) * NT;
    const int arow   = mt * 16 + l15;

    v8f acc[NT];
    #pragma unroll
    for (int nt = 0; nt < NT; ++nt) {
        #pragma unroll
        for (int v = 0; v < 8; ++v) acc[nt][v] = 0.0f;
    }

    const unsigned short* aptr = A + (size_t)arow * lda + hi * 8;
    const unsigned short* bptr = B + (size_t)(ntile0 * 16 + l15) * ldb + hi * 8;

    #pragma unroll 2
    for (int ks = 0; ks < KSTEPS; ++ks) {
        if (ks + 1 < KSTEPS)
            __builtin_prefetch(bptr + (ks + 1) * 32, 0, 1);  // global_prefetch_b8
        v16bf a = load_frag(aptr + ks * 32, 16);             // A layout: +0,+16 elems
        #pragma unroll
        for (int nt = 0; nt < NT; ++nt) {
            const unsigned short* bp = bptr + (size_t)(nt * 16) * ldb + ks * 32;
            v16bf b = load_frag(bp, 16);                     // B layout: +0,+16 elems
            acc[nt] = wmma_bf16(a, b, acc[nt]);
        }
    }

    #pragma unroll
    for (int nt = 0; nt < NT; ++nt) {
        #pragma unroll
        for (int v = 0; v < 8; ++v) {
            int m = v + hi * 8;
            C[(size_t)(mt * 16 + m) * GDIM + (ntile0 + nt) * 16 + l15] = f2bf(acc[nt][v]);
        }
    }
}

// ---------------------------------------------------------------------------
// Recurrent LSTM scan. One block per chain (sent x dir); 512 threads = 16
// waves. Per step: gates[2048] = W_hh @ h + xg[t], via WMMA with A = h
// broadcast into all 16 rows (each lane loads identical LDS chunks), B =
// W_hh^T tile streamed from global (L2-resident). Each wave owns 128 gate
// rows = 8 N-tiles, processed as 4 non-unrolled blocks of 2 tiles so the
// k-loop (unroll x4) pipelines loads without spilling VGPRs.
// ---------------------------------------------------------------------------
__global__ __launch_bounds__(512)
void lstm_scan_kernel(const unsigned short* __restrict__ Whh_base, // [2][2048][512]
                      const unsigned short* __restrict__ xg_base,  // [4][T][2048]
                      unsigned short* __restrict__ hout_base) {    // [2][T][1024]
    const int chain = blockIdx.x;          // 0..3
    const int sent = chain >> 1, dir = chain & 1;
    const unsigned short* W  = Whh_base + (size_t)dir * GDIM * HDIM;
    const unsigned short* xg = xg_base  + (size_t)chain * T_LEN * GDIM;
    unsigned short*     hout = hout_base + (size_t)sent * T_LEN * HID2 + dir * HDIM;

    __shared__ float gates[GDIM];
    __shared__ __align__(16) unsigned short hbuf[HDIM];

    const int tid  = threadIdx.x;
    const int lane = tid & 31;
    const int wave = tid >> 5;             // 0..15
    const int l15  = lane & 15;
    const int hi   = (lane >> 4) & 1;

    hbuf[tid < HDIM ? tid : 0] = 0;        // 512 threads == HDIM
    float cstate = 0.0f;
    __syncthreads();

    for (int step = 0; step < T_LEN; ++step) {
        const int t = dir ? (T_LEN - 1 - step) : step;

        #pragma unroll 1
        for (int ntb = 0; ntb < 4; ++ntb) {      // 2 N-tiles per block
            const int j0 = wave * 128 + ntb * 32;

            // xg loads issued ahead of this block's 32 WMMAs.
            float xgv0 = bf2f(xg[(size_t)t * GDIM + j0 + l15]);
            float xgv1 = bf2f(xg[(size_t)t * GDIM + j0 + 16 + l15]);

            v8f acc0, acc1;
            #pragma unroll
            for (int v = 0; v < 8; ++v) { acc0[v] = 0.0f; acc1[v] = 0.0f; }

            const unsigned short* wrow = W + (size_t)(j0 + l15) * HDIM + hi * 8;

            #pragma unroll 4
            for (int ks = 0; ks < 16; ++ks) {
                // Broadcast-A: every lane loads the same h chunk for its K half.
                v16bf a = load_frag(&hbuf[ks * 32 + hi * 8], 16);
                const unsigned short* bp0 = wrow + ks * 32;
                const unsigned short* bp1 = bp0 + (size_t)16 * HDIM;
                if (ks + 1 < 16)
                    __builtin_prefetch(bp0 + 32, 0, 1);      // global_prefetch_b8
                acc0 = wmma_bf16(a, load_frag(bp0, 16), acc0);
                acc1 = wmma_bf16(a, load_frag(bp1, 16), acc1);
            }

            // D has identical rows; lane n of VGPR0 (lanes 0..15) = y[j0+n].
            if (lane < 16) {
                gates[j0 + l15]      = acc0[0] + xgv0;
                gates[j0 + 16 + l15] = acc1[0] + xgv1;
            }
        }
        __syncthreads();   // all waves done reading hbuf + writing gates

        // Cell update: one hidden unit per thread (torch gate order i,f,g,o).
        {
            const int u = tid;
            float ig = sigmf(gates[u]);
            float fg = sigmf(gates[HDIM + u]);
            float gg = tanhf(gates[2 * HDIM + u]);
            float og = sigmf(gates[3 * HDIM + u]);
            cstate = fg * cstate + ig * gg;
            float h = og * tanhf(cstate);
            unsigned short hb = f2bf(h);
            hout[(size_t)t * HID2 + u] = hb;
            hbuf[u] = hb;
        }
        __syncthreads();   // next step's matmul reads the new hbuf
    }
}

// ---------------------------------------------------------------------------
// feats = h2 @ Wtag^T + btag : [T,16] = [T,1024] x [1024,16]. One wave/block,
// one 16x16 tile, 32 K-steps of WMMA (unroll x4 to bound pressure).
// ---------------------------------------------------------------------------
__global__ __launch_bounds__(32)
void feats_kernel(const unsigned short* __restrict__ h2b,   // [2][T][1024]
                  const unsigned short* __restrict__ wtag,  // [16][1024]
                  const float* __restrict__ btag,
                  float* __restrict__ feats) {              // [2][T][16]
    const int mt = blockIdx.x, sent = blockIdx.y;
    const unsigned short* A = h2b + (size_t)sent * T_LEN * HID2;

    const int lane = threadIdx.x;
    const int l15 = lane & 15;
    const int hi  = (lane >> 4) & 1;
    const int arow = mt * 16 + l15;

    v8f acc;
    #pragma unroll
    for (int v = 0; v < 8; ++v) acc[v] = 0.0f;

    #pragma unroll 4
    for (int ks = 0; ks < HID2 / 32; ++ks) {
        const unsigned short* ap = A    + (size_t)arow * HID2 + ks * 32 + hi * 8;
        const unsigned short* bp = wtag + (size_t)l15  * HID2 + ks * 32 + hi * 8;
        v16bf a = load_frag(ap, 16);
        v16bf b = load_frag(bp, 16);
        acc = wmma_bf16(a, b, acc);
    }

    #pragma unroll
    for (int v = 0; v < 8; ++v) {
        int m = v + hi * 8;
        feats[(size_t)sent * T_LEN * NTAG + (size_t)(mt * 16 + m) * NTAG + l15] =
            acc[v] + btag[l15];
    }
}

// ---------------------------------------------------------------------------
// CRF: forward log-partition + gold score. One wave per sentence; lanes 0..15
// carry alpha[j]. START=14, STOP=15, NEG=-10000.
// ---------------------------------------------------------------------------
__global__ __launch_bounds__(32)
void crf_kernel(const float* __restrict__ feats, const int* __restrict__ pos1,
                const int* __restrict__ pos2, const float* __restrict__ trans,
                float* __restrict__ out) {
    const int sent = blockIdx.x;
    const int lane = threadIdx.x;
    const float* f = feats + (size_t)sent * T_LEN * NTAG;
    const int* tags = sent ? pos2 : pos1;

    __shared__ float ap[NTAG];
    __shared__ float red[32];

    const int lrow = lane & 15;
    float trow[NTAG];
    #pragma unroll
    for (int i = 0; i < NTAG; ++i) trow[i] = trans[lrow * NTAG + i];

    float alpha = (lane == 14) ? 0.0f : -10000.0f;
    for (int t = 0; t < T_LEN; ++t) {
        if (lane < NTAG) ap[lane] = alpha;
        __syncthreads();
        if (lane < NTAG) {
            float m = -3.4e38f;
            #pragma unroll
            for (int i = 0; i < NTAG; ++i) m = fmaxf(m, ap[i] + trow[i]);
            float s = 0.0f;
            #pragma unroll
            for (int i = 0; i < NTAG; ++i) s += __expf(ap[i] + trow[i] - m);
            alpha = m + __logf(s) + f[(size_t)t * NTAG + lane];
        }
        __syncthreads();
    }

    if (lane < NTAG) ap[lane] = alpha + trans[15 * NTAG + lane];
    __syncthreads();
    float term = 0.0f;
    if (lane == 0) {
        float m = -3.4e38f;
        for (int i = 0; i < NTAG; ++i) m = fmaxf(m, ap[i]);
        float s = 0.0f;
        for (int i = 0; i < NTAG; ++i) s += __expf(ap[i] - m);
        term = m + __logf(s);
    }

    float part = 0.0f;
    for (int t = lane; t < T_LEN; t += 32) {
        int tg = tags[t];
        int pr = (t == 0) ? 14 : tags[t - 1];
        part += trans[tg * NTAG + pr] + f[(size_t)t * NTAG + tg];
    }
    red[lane] = part;
    __syncthreads();
    if (lane == 0) {
        float sc = 0.0f;
        for (int i = 0; i < 32; ++i) sc += red[i];
        sc += trans[15 * NTAG + tags[T_LEN - 1]];
        out[1 + sent] = term - sc;
    }
}

// ---------------------------------------------------------------------------
// Final MLP: cat = [h2_s1[T-1], h2_s2[T-1]] (2048) -> relu(Wh) -> sigmoid(Wo)
// ---------------------------------------------------------------------------
__global__ __launch_bounds__(256)
void mlp_kernel(const unsigned short* __restrict__ h2b, const float* __restrict__ Wh,
                const float* __restrict__ bh, const float* __restrict__ Wo,
                const float* __restrict__ bo, float* __restrict__ out) {
    __shared__ float cat[2 * HID2];
    __shared__ float red[256];
    const int tid = threadIdx.x;

    for (int k = tid; k < 2 * HID2; k += 256) {
        int sent = k >> 10, u = k & (HID2 - 1);
        cat[k] = bf2f(h2b[(size_t)sent * T_LEN * HID2 + (size_t)(T_LEN - 1) * HID2 + u]);
    }
    __syncthreads();

    float part = 0.0f;
    for (int r = 0; r < 4; ++r) {
        int u = tid + r * 256;
        float s = bh[u];
        const float* wr = Wh + (size_t)u * (2 * HID2);
        #pragma unroll 4
        for (int k = 0; k < 2 * HID2; ++k) s += cat[k] * wr[k];
        s = fmaxf(s, 0.0f);
        part += s * Wo[u];
    }
    red[tid] = part;
    __syncthreads();
    for (int off = 128; off > 0; off >>= 1) {
        if (tid < off) red[tid] += red[tid + off];
        __syncthreads();
    }
    if (tid == 0) out[0] = 1.0f / (1.0f + __expf(-(red[0] + bo[0])));
}

// ---------------------------------------------------------------------------
// Host launch
// ---------------------------------------------------------------------------
extern "C" void kernel_launch(void* const* d_in, const int* in_sizes, int n_in,
                              void* d_out, int out_size, void* d_ws, size_t ws_size,
                              hipStream_t stream) {
    (void)in_sizes; (void)n_in; (void)out_size; (void)ws_size;

    const int*   s1   = (const int*)d_in[0];
    const int*   s2   = (const int*)d_in[1];
    const int*   pos1 = (const int*)d_in[2];
    const int*   pos2 = (const int*)d_in[3];
    const float* emb  = (const float*)d_in[4];
    const float* Wih0f = (const float*)d_in[5];
    const float* Whh0f = (const float*)d_in[6];
    const float* Wih0b = (const float*)d_in[7];
    const float* Whh0b = (const float*)d_in[8];
    const float* Wih1f = (const float*)d_in[9];
    const float* Whh1f = (const float*)d_in[10];
    const float* Wih1b = (const float*)d_in[11];
    const float* Whh1b = (const float*)d_in[12];
    const float* Wtag  = (const float*)d_in[13];
    const float* btag  = (const float*)d_in[14];
    const float* trans = (const float*)d_in[15];
    const float* Wh    = (const float*)d_in[16];
    const float* bh    = (const float*)d_in[17];
    const float* Wo    = (const float*)d_in[18];
    const float* bo    = (const float*)d_in[19];
    float* out = (float*)d_out;

    // ---- workspace layout (bytes) -----------------------------------------
    char* ws = (char*)d_ws;
    const size_t OFF_XB    = 0x0000000;  // [2][T][64]    bf16   512 KB
    const size_t OFF_WIH0  = 0x0080000;  // [2][2048][64] bf16   512 KB
    const size_t OFF_WHH0  = 0x0100000;  // [2][2048][512]bf16   4 MB
    const size_t OFF_WIH1  = 0x0500000;  // [2][2048][1024]bf16  8 MB
    const size_t OFF_WHH1  = 0x0D00000;  // [2][2048][512]bf16   4 MB
    const size_t OFF_WTAG  = 0x1100000;  // [16][1024]    bf16   32 KB
    const size_t OFF_FEATS = 0x1108000;  // [2][T][16]    f32    256 KB
    const size_t OFF_XG0   = 0x1200000;  // [4][T][2048]  bf16   32 MB
    const size_t OFF_XG1   = 0x3200000;  // [4][T][2048]  bf16   32 MB
    const size_t OFF_H1    = 0x5200000;  // [2][T][1024]  bf16   8 MB
    const size_t OFF_H2    = 0x5A00000;  // [2][T][1024]  bf16   8 MB  (end 0x6200000)

    unsigned short* xb    = (unsigned short*)(ws + OFF_XB);
    unsigned short* wih0  = (unsigned short*)(ws + OFF_WIH0);
    unsigned short* whh0  = (unsigned short*)(ws + OFF_WHH0);
    unsigned short* wih1  = (unsigned short*)(ws + OFF_WIH1);
    unsigned short* whh1  = (unsigned short*)(ws + OFF_WHH1);
    unsigned short* wtag  = (unsigned short*)(ws + OFF_WTAG);
    float*          featsb= (float*)         (ws + OFF_FEATS);
    unsigned short* xg0   = (unsigned short*)(ws + OFF_XG0);
    unsigned short* xg1   = (unsigned short*)(ws + OFF_XG1);
    unsigned short* h1    = (unsigned short*)(ws + OFF_H1);
    unsigned short* h2    = (unsigned short*)(ws + OFF_H2);

    const int CT = 256;
    auto blk = [](int n, int t) { return (n + t - 1) / t; };

    // 1) weight conversion to bf16 (+ zero-pad K where needed)
    cvt_pad_kernel<<<blk(GDIM * EMB_P, CT), CT, 0, stream>>>(Wih0f, wih0,                     GDIM, EMB_D, EMB_P);
    cvt_pad_kernel<<<blk(GDIM * EMB_P, CT), CT, 0, stream>>>(Wih0b, wih0 + GDIM * EMB_P,      GDIM, EMB_D, EMB_P);
    cvt_pad_kernel<<<blk(GDIM * HDIM,  CT), CT, 0, stream>>>(Whh0f, whh0,                     GDIM, HDIM, HDIM);
    cvt_pad_kernel<<<blk(GDIM * HDIM,  CT), CT, 0, stream>>>(Whh0b, whh0 + GDIM * HDIM,       GDIM, HDIM, HDIM);
    cvt_pad_kernel<<<blk(GDIM * HID2,  CT), CT, 0, stream>>>(Wih1f, wih1,                     GDIM, HID2, HID2);
    cvt_pad_kernel<<<blk(GDIM * HID2,  CT), CT, 0, stream>>>(Wih1b, wih1 + GDIM * HID2,       GDIM, HID2, HID2);
    cvt_pad_kernel<<<blk(GDIM * HDIM,  CT), CT, 0, stream>>>(Whh1f, whh1,                     GDIM, HDIM, HDIM);
    cvt_pad_kernel<<<blk(GDIM * HDIM,  CT), CT, 0, stream>>>(Whh1b, whh1 + GDIM * HDIM,       GDIM, HDIM, HDIM);
    cvt_pad_kernel<<<blk(NTAG * HID2,  CT), CT, 0, stream>>>(Wtag,  wtag,                     NTAG, HID2, HID2);

    // 2) embedding gather
    embed_kernel<<<blk(2 * T_LEN * EMB_P, CT), CT, 0, stream>>>(s1, s2, emb, xb);

    // 3) layer-0 input GEMM (K=64 -> 2 K-steps), 4 chains
    gemm_xg_kernel<2><<<dim3(T_LEN / 16, GDIM / (16 * 8 * 4), 4), 256, 0, stream>>>(
        xb, (size_t)T_LEN * EMB_P, EMB_P,
        wih0, (size_t)GDIM * EMB_P, EMB_P,
        xg0, (size_t)T_LEN * GDIM);

    // 4) layer-0 recurrent scan -> h1 (concat fwd|bwd)
    lstm_scan_kernel<<<4, 512, 0, stream>>>(whh0, xg0, h1);

    // 5) layer-1 input GEMM (K=1024 -> 32 K-steps)
    gemm_xg_kernel<32><<<dim3(T_LEN / 16, GDIM / (16 * 8 * 4), 4), 256, 0, stream>>>(
        h1, (size_t)T_LEN * HID2, HID2,
        wih1, (size_t)GDIM * HID2, HID2,
        xg1, (size_t)T_LEN * GDIM);

    // 6) layer-1 recurrent scan -> h2
    lstm_scan_kernel<<<4, 512, 0, stream>>>(whh1, xg1, h2);

    // 7) feats
    feats_kernel<<<dim3(T_LEN / 16, 2), 32, 0, stream>>>(h2, wtag, btag, featsb);

    // 8) CRF losses -> out[1], out[2]
    crf_kernel<<<2, 32, 0, stream>>>(featsb, pos1, pos2, trans, out);

    // 9) similarity head -> out[0]
    mlp_kernel<<<1, 256, 0, stream>>>(h2, Wh, bh, Wo, bo, out);
}
